// FusedMoE_30382598652207
// MI455X (gfx1250) — matrix operand order repaired and checked by reading, not compiled, not run on hardware
//
#include <hip/hip_runtime.h>
#include <stdint.h>

// Problem constants (fixed by the reference).
#define T_DIM 2048
#define H_DIM 2048
#define E_DIM 8
#define I_DIM 5632
#define E_EFF 4        // topk_ids % 4 -> only experts 0..3 routed

typedef __attribute__((ext_vector_type(16))) __bf16 v16bf;
typedef __attribute__((ext_vector_type(2)))  __bf16 v2bf;
typedef __attribute__((ext_vector_type(2)))  float  v2f;
typedef __attribute__((ext_vector_type(8)))  float  v8f;

#define LDT 40         // LDS tile stride in halves (32 + 8 pad -> conflict-free b128)
#define BK  32         // K per iteration == one WMMA K for bf16

// Scheduling fence that blocks only DS ops -> prevents hoisting ALL fragment
// ds_loads to the loop top (VGPR pressure control).
#define DS_FENCE() __builtin_amdgcn_sched_barrier(0x47F)

// ---- fp32 -> bf16 pair via true packed convert (one v_cvt_pk_bf16_f32) ----
__device__ __forceinline__ unsigned pack2(float a, float b) {
  v2f x = {a, b};
  union { v2bf v; unsigned u; } cv;
  cv.v = __builtin_convertvector(x, v2bf);
  return cv.u;
}
__device__ __forceinline__ unsigned short f2bf(float f) {
  return (unsigned short)(pack2(f, f) & 0xFFFFu);
}

// ---- LDS address (addrspace(3) offset) of a generic __shared__ pointer ----
typedef __attribute__((address_space(3))) const unsigned short* lds_csptr;
__device__ __forceinline__ unsigned lds_off(const unsigned short* p) {
  return (unsigned)(unsigned long long)(lds_csptr)p;
}

// ---- CDNA5 async copy: global -> LDS, 32B per lane, no VGPR staging ----
// INST_OFFSET is added to BOTH the memory and LDS address (ISA 15.18.3).
__device__ __forceinline__ void asyncA(unsigned ldsoff, const unsigned short* g) {
  asm volatile(
      "global_load_async_to_lds_b128 %0, %1, off\n\t"
      "global_load_async_to_lds_b128 %0, %1, off offset:16"
      :: "v"(ldsoff), "v"(g) : "memory");
}
__device__ __forceinline__ void wait_async0() {
  asm volatile("s_wait_asynccnt 0x0" ::: "memory");
}

// ---- WMMA fragment loads from LDS (documented CDNA5 layouts) ----
// A 16x32 bf16: lanes 0-15 (M=lane): VGPR0-3 = K0..7, VGPR4-7 = K16..23
//               lanes 16-31 (M=lane-16): K8..15 and K24..31
__device__ __forceinline__ v16bf load_frag_A(const unsigned short* As, int r0, int lane) {
  int row = r0 + (lane & 15);
  int kh  = lane >> 4;                       // 0 or 1
  const unsigned short* p = As + row * LDT + kh * 8;
  v16bf a;
  *(uint4*)&a       = *(const uint4*)p;        // K 0..7  (or 8..15)
  *((uint4*)&a + 1) = *(const uint4*)(p + 16); // K16..23 (or 24..31)
  return a;
}
// B 32x16 bf16 (stored [N][K] in LDS): lanes 0-15 (N=lane) hold K0..7, K16..23,
// lanes 16-31 (N=lane-16) hold K8..15, K24..31; 2 halves per VGPR in K order.
__device__ __forceinline__ v16bf load_frag_B(const unsigned short* Bs, int c0, int lane) {
  int col = c0 + (lane & 15);
  int kh  = lane >> 4;
  const unsigned short* p = Bs + col * LDT + kh * 8;
  v16bf b;
  *(uint4*)&b       = *(const uint4*)p;
  *((uint4*)&b + 1) = *(const uint4*)(p + 16);
  return b;
}

// silu(x) = x * 1/(1+exp(-x)); one v_exp + one v_rcp, no IEEE div sequence.
__device__ __forceinline__ float silu_fast(float x) {
  return x * __builtin_amdgcn_rcpf(1.f + __expf(-x));
}

// ---- B tile fetch (global fp32 -> regs) and stage (cvt -> LDS bf16) ----
// B tile: 128 x 32 fp32; 256 threads: 2 per row, 16 elements each.
__device__ __forceinline__ void fetchB(const float* bsrc, int k0, float4 f[4]) {
  f[0] = *(const float4*)(bsrc + k0);
  f[1] = *(const float4*)(bsrc + k0 + 4);
  f[2] = *(const float4*)(bsrc + k0 + 8);
  f[3] = *(const float4*)(bsrc + k0 + 12);
}
__device__ __forceinline__ void stageB(unsigned short* Bsd, int grow, int gcol,
                                       const float4 f[4]) {
  uint4 p0, p1;
  p0.x = pack2(f[0].x, f[0].y); p0.y = pack2(f[0].z, f[0].w);
  p0.z = pack2(f[1].x, f[1].y); p0.w = pack2(f[1].z, f[1].w);
  p1.x = pack2(f[2].x, f[2].y); p1.y = pack2(f[2].z, f[2].w);
  p1.z = pack2(f[3].x, f[3].y); p1.w = pack2(f[3].z, f[3].w);
  uint4* sB = (uint4*)(Bsd + grow * LDT + gcol);
  sB[0] = p0; sB[1] = p1;
}

// ---------------- setup kernels ----------------
__global__ void zero_f32(float* __restrict__ p, int n) {
  int i = blockIdx.x * blockDim.x + threadIdx.x;
  if (i < n) p[i] = 0.f;
}

__global__ void cvt_hidden(const float* __restrict__ h, unsigned short* __restrict__ hb) {
  int i = (blockIdx.x * blockDim.x + threadIdx.x) * 4;
  float4 v = *(const float4*)(h + i);
  uint2 p;
  p.x = pack2(v.x, v.y);
  p.y = pack2(v.z, v.w);
  *(uint2*)(hb + i) = p;
}

__global__ void route_kernel(const int* __restrict__ ids, const float* __restrict__ w,
                             float* __restrict__ route) {
  int t = blockIdx.x * blockDim.x + threadIdx.x;
  if (t >= T_DIM) return;
  float r[E_EFF] = {0.f, 0.f, 0.f, 0.f};
  for (int k = 0; k < 2; ++k) {
    int e = ids[t * 2 + k] & 3;       // == % 4 (ids nonnegative)
    r[e] += w[t * 2 + k];
  }
  for (int e = 0; e < E_EFF; ++e) route[(size_t)e * T_DIM + t] = r[e];
}

// ---------------- GEMM1 + SwiGLU: act = silu(x W1^T) * (x W3^T) ----------------
// Block tile 128(M) x 64(N), 8 waves as 4x2, wave tile 32x32 for BOTH g1 and g3.
__device__ __forceinline__ void g1_compute(const unsigned short* Asd, const unsigned short* Bsd,
                                           int waveM, int waveN, int lane,
                                           v8f acc1[2][2], v8f acc3[2][2]) {
  v16bf af[2];
#pragma unroll
  for (int mt = 0; mt < 2; ++mt)
    af[mt] = load_frag_A(Asd, waveM * 32 + mt * 16, lane);
#pragma unroll
  for (int nt = 0; nt < 2; ++nt) {
    v16bf b1 = load_frag_B(Bsd, waveN * 32 + nt * 16, lane);
    v16bf b3 = load_frag_B(Bsd, 64 + waveN * 32 + nt * 16, lane);
#pragma unroll
    for (int mt = 0; mt < 2; ++mt) {
      acc1[mt][nt] = __builtin_amdgcn_wmma_f32_16x16x32_bf16(
          false, af[mt], false, b1, (short)0, acc1[mt][nt], false, false);
      acc3[mt][nt] = __builtin_amdgcn_wmma_f32_16x16x32_bf16(
          false, af[mt], false, b3, (short)0, acc3[mt][nt], false, false);
    }
    DS_FENCE();   // keep next nt's B-fragment ds_loads below -> cap live fragments
  }
}

__global__ __launch_bounds__(256, 1) void moe_gemm1(
    const unsigned short* __restrict__ hidB,   // [T][H] bf16
    const float* __restrict__ w13,             // [E][2I][H] fp32
    unsigned short* __restrict__ act,          // [T][I] bf16 (this expert)
    int e)
{
  __shared__ unsigned short As[2][128 * LDT];
  __shared__ unsigned short Bs[2][128 * LDT]; // rows 0-63: W1 cols, 64-127: W3 cols

  const int tid   = threadIdx.x;
  const int lane  = tid & 31;
  const int wid   = tid >> 5;
  const int waveM = wid & 3;                   // 4 along M
  const int waveN = wid >> 2;                  // 2 along N
  const int m0 = blockIdx.y * 128;
  const int n0 = blockIdx.x * 64;

  const int grow = tid >> 1;
  const int gcol = (tid & 1) * 16;
  const unsigned short* asrc = hidB + (size_t)(m0 + grow) * H_DIM + gcol;
  const size_t w13row = (grow < 64) ? (size_t)(n0 + grow)
                                    : (size_t)(I_DIM + n0 + (grow - 64));
  const float* bsrc = w13 + ((size_t)e * 2 * I_DIM + w13row) * H_DIM + gcol;

  const unsigned ldsA0 = lds_off(As[0] + grow * LDT + gcol);
  const unsigned ldsA1 = lds_off(As[1] + grow * LDT + gcol);

  v8f zf = {};
  v8f acc1[2][2], acc3[2][2];
  for (int i = 0; i < 2; ++i)
    for (int j = 0; j < 2; ++j) { acc1[i][j] = zf; acc3[i][j] = zf; }

  float4 freg[4];
  asyncA(ldsA0, asrc);
  fetchB(bsrc, 0, freg);
  stageB(Bs[0], grow, gcol, freg);
  wait_async0();
  __syncthreads();

  // two-stage software pipeline: A via async DMA to LDS, B via regs (cvt) after WMMAs
  for (int k0 = 0; k0 < H_DIM; k0 += 2 * BK) {
    asyncA(ldsA1, asrc + k0 + BK);
    fetchB(bsrc, k0 + BK, freg);
    g1_compute(As[0], Bs[0], waveM, waveN, lane, acc1, acc3);
    stageB(Bs[1], grow, gcol, freg);
    wait_async0();
    __syncthreads();

    const bool more = (k0 + 2 * BK) < H_DIM;
    if (more) {
      asyncA(ldsA0, asrc + k0 + 2 * BK);
      fetchB(bsrc, k0 + 2 * BK, freg);
    }
    g1_compute(As[1], Bs[1], waveM, waveN, lane, acc1, acc3);
    if (more) stageB(Bs[0], grow, gcol, freg);
    wait_async0();
    __syncthreads();
  }

  // SwiGLU fused epilogue -> bf16 act (fully unrolled: constant vector indices)
  const int lrow = 8 * (lane >> 4);
  const int lcol = lane & 15;
#pragma unroll
  for (int mt = 0; mt < 2; ++mt)
#pragma unroll
    for (int nt = 0; nt < 2; ++nt) {
      int rbase = m0 + waveM * 32 + mt * 16 + lrow;
      int c     = n0 + waveN * 32 + nt * 16 + lcol;
#pragma unroll
      for (int j = 0; j < 8; ++j) {
        float g1 = acc1[mt][nt][j];
        float g3 = acc3[mt][nt][j];
        float a  = silu_fast(g1) * g3;
        act[(size_t)(rbase + j) * I_DIM + c] = f2bf(a);
      }
    }
}

// ---------------- GEMM2 + routed accumulate: out += route[e]*(act W2^T) ----------------
// Block tile 128x128, 8 waves as 2x4, wave tile 64(M) x 32(N).
__device__ __forceinline__ void g2_compute(const unsigned short* Asd, const unsigned short* Bsd,
                                           int waveM, int waveN, int lane, v8f acc[4][2]) {
  v16bf bfr[2];
#pragma unroll
  for (int nt = 0; nt < 2; ++nt)
    bfr[nt] = load_frag_B(Bsd, waveN * 32 + nt * 16, lane);
#pragma unroll
  for (int mt = 0; mt < 4; ++mt) {
    v16bf af = load_frag_A(Asd, waveM * 64 + mt * 16, lane);
#pragma unroll
    for (int nt = 0; nt < 2; ++nt)
      acc[mt][nt] = __builtin_amdgcn_wmma_f32_16x16x32_bf16(
          false, af, false, bfr[nt], (short)0, acc[mt][nt], false, false);
    DS_FENCE();   // keep next mt's A-fragment ds_loads below -> cap live fragments
  }
}

__global__ __launch_bounds__(256, 1) void moe_gemm2(
    const unsigned short* __restrict__ act,    // [T][I] bf16
    const float* __restrict__ w2,              // [E][H][I] fp32
    const float* __restrict__ route,           // [4][T]
    float* __restrict__ out,                   // [T][H] fp32
    int e)
{
  __shared__ unsigned short As[2][128 * LDT];
  __shared__ unsigned short Bs[2][128 * LDT];

  const int tid   = threadIdx.x;
  const int lane  = tid & 31;
  const int wid   = tid >> 5;
  const int waveM = wid & 1;                   // 2 along M (x64)
  const int waveN = wid >> 1;                  // 4 along N (x32)
  const int m0 = blockIdx.y * 128;
  const int n0 = blockIdx.x * 128;

  const int grow = tid >> 1;
  const int gcol = (tid & 1) * 16;
  const unsigned short* asrc = act + (size_t)(m0 + grow) * I_DIM + gcol;
  const float* bsrc = w2 + ((size_t)e * H_DIM + (n0 + grow)) * I_DIM + gcol;

  const unsigned ldsA0 = lds_off(As[0] + grow * LDT + gcol);
  const unsigned ldsA1 = lds_off(As[1] + grow * LDT + gcol);

  v8f zf = {};
  v8f acc[4][2];
  for (int i = 0; i < 4; ++i)
    for (int j = 0; j < 2; ++j) acc[i][j] = zf;

  float4 freg[4];
  asyncA(ldsA0, asrc);
  fetchB(bsrc, 0, freg);
  stageB(Bs[0], grow, gcol, freg);
  wait_async0();
  __syncthreads();

  for (int k0 = 0; k0 < I_DIM; k0 += 2 * BK) {
    asyncA(ldsA1, asrc + k0 + BK);
    fetchB(bsrc, k0 + BK, freg);
    g2_compute(As[0], Bs[0], waveM, waveN, lane, acc);
    stageB(Bs[1], grow, gcol, freg);
    wait_async0();
    __syncthreads();

    const bool more = (k0 + 2 * BK) < I_DIM;
    if (more) {
      asyncA(ldsA0, asrc + k0 + 2 * BK);
      fetchB(bsrc, k0 + 2 * BK, freg);
    }
    g2_compute(As[1], Bs[1], waveM, waveN, lane, acc);
    if (more) stageB(Bs[0], grow, gcol, freg);
    wait_async0();
    __syncthreads();
  }

  // routed accumulate into out (exclusive tiles; experts sequential -> no race)
  const int lrow = 8 * (lane >> 4);
  const int lcol = lane & 15;
#pragma unroll
  for (int mt = 0; mt < 4; ++mt)
#pragma unroll
    for (int nt = 0; nt < 2; ++nt) {
      int rbase = m0 + waveM * 64 + mt * 16 + lrow;
      int c     = n0 + waveN * 32 + nt * 16 + lcol;
#pragma unroll
      for (int j = 0; j < 8; ++j) {
        int row = rbase + j;
        float r = route[(size_t)e * T_DIM + row];
        size_t o = (size_t)row * H_DIM + c;
        out[o] += r * acc[mt][nt][j];
      }
    }
}

extern "C" void kernel_launch(void* const* d_in, const int* in_sizes, int n_in,
                              void* d_out, int out_size, void* d_ws, size_t ws_size,
                              hipStream_t stream) {
  const float* hidden = (const float*)d_in[0];   // [T][H]
  const float* w13    = (const float*)d_in[1];   // [E][2I][H]
  const float* w2     = (const float*)d_in[2];   // [E][H][I]
  const float* tkw    = (const float*)d_in[3];   // [T][2]
  const int*   tki    = (const int*)d_in[4];     // [T][2]
  float* out = (float*)d_out;

  // workspace layout: hid_bf16 (8MB) | act_bf16 (23MB, reused per expert) | route (32KB)
  unsigned short* hidB  = (unsigned short*)d_ws;
  unsigned short* act   = hidB + (size_t)T_DIM * H_DIM;
  float*          route = (float*)(act + (size_t)T_DIM * I_DIM);

  zero_f32<<<(T_DIM * H_DIM + 255) / 256, 256, 0, stream>>>(out, T_DIM * H_DIM);
  cvt_hidden<<<(T_DIM * H_DIM / 4 + 255) / 256, 256, 0, stream>>>(hidden, hidB);
  route_kernel<<<(T_DIM + 255) / 256, 256, 0, stream>>>(tki, tkw, route);

  for (int e = 0; e < E_EFF; ++e) {
    moe_gemm1<<<dim3(I_DIM / 64, T_DIM / 128), 256, 0, stream>>>(hidB, w13, act, e);
    moe_gemm2<<<dim3(H_DIM / 128, T_DIM / 128), 256, 0, stream>>>(act, w2, route, out, e);
  }
}
